// TextEncoder_63299228008788
// MI455X (gfx1250) — matrix operand order repaired and checked by reading, not compiled, not run on hardware
//
#include <hip/hip_runtime.h>
#include <hip/hip_bf16.h>
#include <math.h>

#define BB   256
#define SS   96
#define DD   500
#define DP   512
#define HH   1024
#define H3   3072
#define RNN  2048
#define FIN  3584
#define NOUT 2048
#define MS   (BB * SS)   /* 24576 */

typedef __attribute__((ext_vector_type(16))) _Float16 v16h;
typedef __attribute__((ext_vector_type(8)))  _Float16 v8h;
typedef __attribute__((ext_vector_type(8)))  float    v8f;

__device__ __forceinline__ v8f wmma16(v16h a, v16h b, v8f c) {
  // D = A(16x32,f16) * B(32x16,f16) + C(16x16,f32)
  return __builtin_amdgcn_wmma_f32_16x16x32_f16(false, a, false, b, (short)0, c,
                                                false, false);
}

// A fragment 16x32: lane holds row M=lane&15; K chunks [klo,klo+8) and [16+klo,16+klo+8)
__device__ __forceinline__ v16h load_a_frag(const _Float16* tile, int lda, int lane) {
  const int m   = lane & 15;
  const int klo = (lane < 16) ? 0 : 8;
  const _Float16* p = tile + (size_t)m * lda + klo;
  v8h lo = *(const v8h*)(p);
  v8h hi = *(const v8h*)(p + 16);
  v16h a;
#pragma unroll
  for (int i = 0; i < 8; ++i) { a[i] = lo[i]; a[i + 8] = hi[i]; }
  return a;
}

// B fragment 32x16 from Bt[N][K] (row-major over K): lane holds column N=lane&15,
// K chunks [kb, kb+8) and [16+kb, 16+kb+8), kb = 0 or 8.
__device__ __forceinline__ v16h load_b_frag(const _Float16* tile, int ldb, int lane) {
  const int n  = lane & 15;
  const int kb = (lane < 16) ? 0 : 8;
  const _Float16* p = tile + (size_t)n * ldb + kb;
  v8h lo = *(const v8h*)(p);
  v8h hi = *(const v8h*)(p + 16);
  v16h b;
#pragma unroll
  for (int i = 0; i < 8; ++i) { b[i] = lo[i]; b[i + 8] = hi[i]; }
  return b;
}

// ---------------------------------------------------------------- utility ----
__global__ void k_zero_f32(float* p, int n) {
  int i = blockIdx.x * blockDim.x + threadIdx.x;
  if (i < n) p[i] = 0.0f;
}
__global__ void k_zero_f16(_Float16* p, int n) {
  int i = blockIdx.x * blockDim.x + threadIdx.x;
  if (i < n) p[i] = (_Float16)0.0f;
}
__global__ void k_cvt_pad(const float* __restrict__ src, _Float16* __restrict__ dst,
                          int rows, int cs, int cd) {
  int i = blockIdx.x * blockDim.x + threadIdx.x;
  if (i >= rows * cd) return;
  int r = i / cd, c = i - r * cd;
  dst[i] = (c < cs) ? (_Float16)src[(size_t)r * cs + c] : (_Float16)0.0f;
}
__global__ void k_embed(const int* __restrict__ wids, const float* __restrict__ emb,
                        _Float16* __restrict__ xh) {
  int i = blockIdx.x * blockDim.x + threadIdx.x;
  if (i >= MS * DP) return;
  int d = i & (DP - 1);
  int r = i >> 9;  // /512
  int wid = wids[r];
  xh[i] = (d < DD) ? (_Float16)emb[(size_t)wid * DD + d] : (_Float16)0.0f;
}

// ------------------------------------------------- generic GEMM, f16 out ----
// C[M][N](f16) = A[M][K](f16) * Bt[N][K]^T + bias[N]
// Per wave: 64x32 output (4 M-tiles x 2 N-tiles). grid(M/64, N/128), block 128.
__global__ void k_gemm_bias_f16(const _Float16* __restrict__ A,
                                const _Float16* __restrict__ Bt,
                                const float* __restrict__ bias,
                                _Float16* __restrict__ C, int N, int K) {
  const int lane = threadIdx.x & 31;
  const int wave = threadIdx.x >> 5;
  const int m0 = blockIdx.x * 64;
  const int n0 = (blockIdx.y * 4 + wave) * 32;
  v8f acc[4][2] = {};
  for (int kk = 0; kk < K; kk += 32) {
    v16h bf[2];
#pragma unroll
    for (int j = 0; j < 2; ++j)
      bf[j] = load_b_frag(Bt + (size_t)(n0 + 16 * j) * K + kk, K, lane);
#pragma unroll
    for (int i = 0; i < 4; ++i) {
      v16h a = load_a_frag(A + (size_t)(m0 + 16 * i) * K + kk, K, lane);
#pragma unroll
      for (int j = 0; j < 2; ++j) acc[i][j] = wmma16(a, bf[j], acc[i][j]);
    }
  }
  const int rbase = (lane < 16) ? 0 : 8;
#pragma unroll
  for (int j = 0; j < 2; ++j) {
    const int col = n0 + 16 * j + (lane & 15);
    const float bv = bias[col];
#pragma unroll
    for (int i = 0; i < 4; ++i)
#pragma unroll
      for (int r = 0; r < 8; ++r)
        C[(size_t)(m0 + 16 * i + rbase + r) * N + col] = (_Float16)(acc[i][j][r] + bv);
  }
}

// ------------------------------------------------------------- GRU step ----
// One timestep of one direction. hg = h @ w_hh^T (+b_hh), gates, update.
// Per wave: 32 batch rows x 16 h-cols x 3 gates (A fragment shared by gates).
// grid(8, 16), block 128. hin/hout are ping-pong f16 state buffers.
__global__ void k_gru_step(const _Float16* __restrict__ hin,
                           const _Float16* __restrict__ whh,   // [3H][H] f16
                           const float* __restrict__ bhh,      // [3H]
                           const _Float16* __restrict__ xg,    // [MS][3H] f16
                           const int* __restrict__ lengths,
                           float* __restrict__ h32,
                           _Float16* __restrict__ hout,
                           _Float16* __restrict__ seq,         // [B][S][2H] f16
                           float* __restrict__ pool,           // [B][2H]
                           int t, int dir) {
  const int lane = threadIdx.x & 31;
  const int wave = threadIdx.x >> 5;
  const int b0 = blockIdx.x * 32;          // 2 M tiles
  const int h0 = (blockIdx.y * 4 + wave) * 16;

  v8f ar[2] = {}, az[2] = {}, an[2] = {};
  const _Float16* br0 = whh + (size_t)(h0) * HH;
  const _Float16* bz0 = whh + (size_t)(h0 + HH) * HH;
  const _Float16* bn0 = whh + (size_t)(h0 + 2 * HH) * HH;
  for (int kk = 0; kk < HH; kk += 32) {
    v16h br = load_b_frag(br0 + kk, HH, lane);
    v16h bz = load_b_frag(bz0 + kk, HH, lane);
    v16h bn = load_b_frag(bn0 + kk, HH, lane);
#pragma unroll
    for (int i = 0; i < 2; ++i) {
      v16h a = load_a_frag(hin + (size_t)(b0 + 16 * i) * HH + kk, HH, lane);
      ar[i] = wmma16(a, br, ar[i]);
      az[i] = wmma16(a, bz, az[i]);
      an[i] = wmma16(a, bn, an[i]);
    }
  }
  const int col = h0 + (lane & 15);
  const float brv = bhh[col], bzv = bhh[col + HH], bnv = bhh[col + 2 * HH];
  const int rbase = (lane < 16) ? 0 : 8;
#pragma unroll
  for (int i = 0; i < 2; ++i) {
#pragma unroll
    for (int r = 0; r < 8; ++r) {
      const int b = b0 + 16 * i + rbase + r;
      const size_t xrow = (size_t)(b * SS + t) * H3;
      const float xr = (float)xg[xrow + col];
      const float xz = (float)xg[xrow + col + HH];
      const float xn = (float)xg[xrow + col + 2 * HH];
      const float rg = 1.0f / (1.0f + __expf(-(xr + ar[i][r] + brv)));
      const float zg = 1.0f / (1.0f + __expf(-(xz + az[i][r] + bzv)));
      const float ng = tanhf(xn + rg * (an[i][r] + bnv));
      const size_t hidx = (size_t)b * HH + col;
      const float hp = h32[hidx];
      const float hnew = (1.0f - zg) * ng + zg * hp;
      const bool m = (t < lengths[b]);
      const float hs = m ? hnew : hp;
      h32[hidx] = hs;
      hout[hidx] = (_Float16)hs;
      const float ov = m ? hs : 0.0f;
      seq[((size_t)b * SS + t) * RNN + dir * HH + col] = (_Float16)ov;
      pool[(size_t)b * RNN + dir * HH + col] += ov;
    }
  }
}

// ------------------------------------------------------------ pool / div ----
__global__ void k_pool_div(const float* __restrict__ pool, const int* __restrict__ len,
                           _Float16* __restrict__ feat) {
  int i = blockIdx.x * blockDim.x + threadIdx.x;
  if (i >= BB * RNN) return;
  int b = i / RNN, j = i - b * RNN;
  feat[(size_t)b * FIN + j] = (_Float16)(pool[i] / (float)len[b]);
}

// ------------------------------------------------------------------ conv ----
// conv ws over time, padding (ws-1,ws-1), relu(+bias), max over time.
// Per wave: 32 time rows x 32 channels (2x2 tiles). grid(B, 4), block 128.
__device__ __forceinline__ v16h load_a_conv(const _Float16* seqb, int trow, int koff,
                                            int klo) {
  v16h a;
  if (trow >= 0 && trow < SS) {
    const _Float16* p = seqb + (size_t)trow * RNN + koff + klo;
    v8h lo = *(const v8h*)(p);
    v8h hi = *(const v8h*)(p + 16);
#pragma unroll
    for (int i = 0; i < 8; ++i) { a[i] = lo[i]; a[i + 8] = hi[i]; }
  } else {
#pragma unroll
    for (int i = 0; i < 16; ++i) a[i] = (_Float16)0.0f;
  }
  return a;
}

__global__ void k_conv(const _Float16* __restrict__ seq,   // [B][S][2H] f16
                       const _Float16* __restrict__ W,     // [512][ws*2H] f16
                       const float* __restrict__ bias,     // [512]
                       _Float16* __restrict__ feat, int ws, int featOff) {
  const int lane = threadIdx.x & 31;
  const int wave = threadIdx.x >> 5;
  const int b = blockIdx.x;
  const int n0 = (blockIdx.y * 4 + wave) * 32;  // 2 N tiles
  const int Kc = ws * RNN;
  const int Tout = SS + ws - 1;
  const int m32 = (Tout + 31) / 32;
  const int rbase = (lane < 16) ? 0 : 8;
  const int klo = (lane < 16) ? 0 : 8;
  const _Float16* seqb = seq + (size_t)b * SS * RNN;
  const float bv0 = bias[n0 + (lane & 15)];
  const float bv1 = bias[n0 + 16 + (lane & 15)];
  float mx[2] = {0.0f, 0.0f};

  for (int mt = 0; mt < m32; ++mt) {
    const int t0 = mt * 32;
    v8f acc[2][2] = {};
    for (int j = 0; j < ws; ++j) {
      const int tl0 = t0 + (lane & 15) - (ws - 1) + j;        // M tile 0 per-lane row
      const int tl1 = tl0 + 16;                               // M tile 1 per-lane row
      for (int kk = 0; kk < RNN; kk += 32) {
        v16h bf0 = load_b_frag(W + (size_t)n0 * Kc + j * RNN + kk, Kc, lane);
        v16h bf1 = load_b_frag(W + (size_t)(n0 + 16) * Kc + j * RNN + kk, Kc, lane);
        v16h a0 = load_a_conv(seqb, tl0, kk, klo);
        v16h a1 = load_a_conv(seqb, tl1, kk, klo);
        acc[0][0] = wmma16(a0, bf0, acc[0][0]);
        acc[0][1] = wmma16(a0, bf1, acc[0][1]);
        acc[1][0] = wmma16(a1, bf0, acc[1][0]);
        acc[1][1] = wmma16(a1, bf1, acc[1][1]);
      }
    }
#pragma unroll
    for (int i = 0; i < 2; ++i)
#pragma unroll
      for (int r = 0; r < 8; ++r) {
        const int tpos = t0 + 16 * i + rbase + r;
        if (tpos < Tout) {
          float v0 = acc[i][0][r] + bv0;
          float v1 = acc[i][1][r] + bv1;
          v0 = v0 > 0.0f ? v0 : 0.0f;
          v1 = v1 > 0.0f ? v1 : 0.0f;
          mx[0] = fmaxf(mx[0], v0);
          mx[1] = fmaxf(mx[1], v1);
        }
      }
  }
  mx[0] = fmaxf(mx[0], __shfl_xor(mx[0], 16, 32));  // merge row-halves (same N)
  mx[1] = fmaxf(mx[1], __shfl_xor(mx[1], 16, 32));
  if (lane < 16) {
    feat[(size_t)b * FIN + featOff + n0 + lane]      = (_Float16)mx[0];
    feat[(size_t)b * FIN + featOff + n0 + 16 + lane] = (_Float16)mx[1];
  }
}

// -------------------------------------------------------------------- FC ----
// outpre = relu(feat @ fc_w^T + fc_b); sumsq[b] += v*v.
// Per wave: 32x32 (2x2 tiles). grid(8, 16), block 128.
__global__ void k_fc(const _Float16* __restrict__ A, const _Float16* __restrict__ Bt,
                     const float* __restrict__ bias, float* __restrict__ outpre,
                     float* __restrict__ sumsq) {
  const int lane = threadIdx.x & 31;
  const int wave = threadIdx.x >> 5;
  const int m0 = blockIdx.x * 32;
  const int n0 = (blockIdx.y * 4 + wave) * 32;
  v8f acc[2][2] = {};
  for (int kk = 0; kk < FIN; kk += 32) {
    v16h bf[2];
#pragma unroll
    for (int j = 0; j < 2; ++j)
      bf[j] = load_b_frag(Bt + (size_t)(n0 + 16 * j) * FIN + kk, FIN, lane);
#pragma unroll
    for (int i = 0; i < 2; ++i) {
      v16h a = load_a_frag(A + (size_t)(m0 + 16 * i) * FIN + kk, FIN, lane);
#pragma unroll
      for (int j = 0; j < 2; ++j) acc[i][j] = wmma16(a, bf[j], acc[i][j]);
    }
  }
  const int rbase = (lane < 16) ? 0 : 8;
#pragma unroll
  for (int j = 0; j < 2; ++j) {
    const int col = n0 + 16 * j + (lane & 15);
    const float bv = bias[col];
#pragma unroll
    for (int i = 0; i < 2; ++i)
#pragma unroll
      for (int r = 0; r < 8; ++r) {
        const int row = m0 + 16 * i + rbase + r;
        float v = acc[i][j][r] + bv;
        v = v > 0.0f ? v : 0.0f;
        outpre[(size_t)row * NOUT + col] = v;
        atomicAdd(&sumsq[row], v * v);
      }
  }
}

__global__ void k_norm(const float* __restrict__ outpre, const float* __restrict__ sumsq,
                       float* __restrict__ out) {
  int i = blockIdx.x * blockDim.x + threadIdx.x;
  if (i >= BB * NOUT) return;
  int b = i / NOUT;
  out[i] = outpre[i] / sqrtf(sumsq[b]);
}

// ----------------------------------------------------------------- launch ----
extern "C" void kernel_launch(void* const* d_in, const int* in_sizes, int n_in,
                              void* d_out, int out_size, void* d_ws, size_t ws_size,
                              hipStream_t stream) {
  (void)in_sizes; (void)n_in; (void)out_size; (void)ws_size;
  const int*   cap     = (const int*)d_in[0];
  const int*   lens    = (const int*)d_in[1];
  const float* embed_w = (const float*)d_in[2];
  const float* w_ih_f  = (const float*)d_in[3];
  const float* w_hh_f  = (const float*)d_in[4];
  const float* b_ih_f  = (const float*)d_in[5];
  const float* b_hh_f  = (const float*)d_in[6];
  const float* w_ih_b  = (const float*)d_in[7];
  const float* w_hh_b  = (const float*)d_in[8];
  const float* b_ih_b  = (const float*)d_in[9];
  const float* b_hh_b  = (const float*)d_in[10];
  const float* conv_w2 = (const float*)d_in[11];
  const float* conv_b2 = (const float*)d_in[12];
  const float* conv_w3 = (const float*)d_in[13];
  const float* conv_b3 = (const float*)d_in[14];
  const float* conv_w4 = (const float*)d_in[15];
  const float* conv_b4 = (const float*)d_in[16];
  const float* fc_w    = (const float*)d_in[17];
  const float* fc_b    = (const float*)d_in[18];
  float* out = (float*)d_out;

  char* wp = (char*)d_ws;
  auto alloc = [&](size_t bytes) -> void* {
    void* p = (void*)wp;
    wp += (bytes + 255) & ~(size_t)255;
    return p;
  };
  _Float16* xh    = (_Float16*)alloc((size_t)MS * DP * 2);
  _Float16* wihf  = (_Float16*)alloc((size_t)H3 * DP * 2);
  _Float16* whhf  = (_Float16*)alloc((size_t)H3 * HH * 2);
  _Float16* wihb  = (_Float16*)alloc((size_t)H3 * DP * 2);
  _Float16* whhb  = (_Float16*)alloc((size_t)H3 * HH * 2);
  _Float16* fcw   = (_Float16*)alloc((size_t)NOUT * FIN * 2);
  _Float16* cw2   = (_Float16*)alloc((size_t)512 * 2 * RNN * 2);
  _Float16* cw3   = (_Float16*)alloc((size_t)512 * 3 * RNN * 2);
  _Float16* cw4   = (_Float16*)alloc((size_t)512 * 4 * RNN * 2);
  _Float16* xg    = (_Float16*)alloc((size_t)MS * H3 * 2);      // reused per direction
  _Float16* seq   = (_Float16*)alloc((size_t)BB * SS * RNN * 2);
  float*    h32   = (float*)alloc((size_t)BB * HH * 4);
  _Float16* h16a  = (_Float16*)alloc((size_t)BB * HH * 2);
  _Float16* h16b  = (_Float16*)alloc((size_t)BB * HH * 2);
  float*    pool  = (float*)alloc((size_t)BB * RNN * 4);
  _Float16* feat  = (_Float16*)alloc((size_t)BB * FIN * 2);
  float*    opre  = (float*)alloc((size_t)BB * NOUT * 4);
  float*    ssq   = (float*)alloc((size_t)BB * 4);

  const int T = 256;
  auto gz = [&](int n) { return dim3((n + T - 1) / T); };

  // stage inputs as f16
  k_embed<<<gz(MS * DP), T, 0, stream>>>(cap, embed_w, xh);
  k_cvt_pad<<<gz(H3 * DP), T, 0, stream>>>(w_ih_f, wihf, H3, DD, DP);
  k_cvt_pad<<<gz(H3 * HH), T, 0, stream>>>(w_hh_f, whhf, H3, HH, HH);
  k_cvt_pad<<<gz(H3 * DP), T, 0, stream>>>(w_ih_b, wihb, H3, DD, DP);
  k_cvt_pad<<<gz(H3 * HH), T, 0, stream>>>(w_hh_b, whhb, H3, HH, HH);
  k_cvt_pad<<<gz(NOUT * FIN), T, 0, stream>>>(fc_w, fcw, NOUT, FIN, FIN);
  k_cvt_pad<<<gz(512 * 2 * RNN), T, 0, stream>>>(conv_w2, cw2, 512, 2 * RNN, 2 * RNN);
  k_cvt_pad<<<gz(512 * 3 * RNN), T, 0, stream>>>(conv_w3, cw3, 512, 3 * RNN, 3 * RNN);
  k_cvt_pad<<<gz(512 * 4 * RNN), T, 0, stream>>>(conv_w4, cw4, 512, 4 * RNN, 4 * RNN);

  k_zero_f32<<<gz(BB * RNN), T, 0, stream>>>(pool, BB * RNN);
  k_zero_f32<<<gz(BB), T, 0, stream>>>(ssq, BB);

  // ---- forward direction ----
  k_gemm_bias_f16<<<dim3(MS / 64, H3 / 128), 128, 0, stream>>>(xh, wihf, b_ih_f, xg,
                                                               H3, DP);
  k_zero_f32<<<gz(BB * HH), T, 0, stream>>>(h32, BB * HH);
  k_zero_f16<<<gz(BB * HH), T, 0, stream>>>(h16a, BB * HH);
  {
    _Float16* hin = h16a; _Float16* hout = h16b;
    for (int t = 0; t < SS; ++t) {
      k_gru_step<<<dim3(BB / 32, 16), 128, 0, stream>>>(hin, whhf, b_hh_f, xg, lens,
                                                        h32, hout, seq, pool, t, 0);
      _Float16* tmp = hin; hin = hout; hout = tmp;
    }
  }
  // ---- backward direction (reuses xg buffer) ----
  k_gemm_bias_f16<<<dim3(MS / 64, H3 / 128), 128, 0, stream>>>(xh, wihb, b_ih_b, xg,
                                                               H3, DP);
  k_zero_f32<<<gz(BB * HH), T, 0, stream>>>(h32, BB * HH);
  k_zero_f16<<<gz(BB * HH), T, 0, stream>>>(h16a, BB * HH);
  {
    _Float16* hin = h16a; _Float16* hout = h16b;
    for (int t = SS - 1; t >= 0; --t) {
      k_gru_step<<<dim3(BB / 32, 16), 128, 0, stream>>>(hin, whhb, b_hh_b, xg, lens,
                                                        h32, hout, seq, pool, t, 1);
      _Float16* tmp = hin; hin = hout; hout = tmp;
    }
  }

  // pooled GRU features -> feat[:, 0:2048]
  k_pool_div<<<gz(BB * RNN), T, 0, stream>>>(pool, lens, feat);

  // convs -> feat[:, 2048:3584]
  k_conv<<<dim3(BB, 4), 128, 0, stream>>>(seq, cw2, conv_b2, feat, 2, RNN + 0);
  k_conv<<<dim3(BB, 4), 128, 0, stream>>>(seq, cw3, conv_b3, feat, 3, RNN + 512);
  k_conv<<<dim3(BB, 4), 128, 0, stream>>>(seq, cw4, conv_b4, feat, 4, RNN + 1024);

  // FC + relu + L2 normalize
  k_fc<<<dim3(BB / 32, NOUT / 128), 128, 0, stream>>>(feat, fcw, fc_b, opre, ssq);
  k_norm<<<gz(BB * NOUT), T, 0, stream>>>(opre, ssq, out);
}
